// NT_Xent_70437463654455
// MI455X (gfx1250) — matrix operand order repaired and checked by reading, not compiled, not run on hardware
//
#include <hip/hip_runtime.h>
#include <hip/hip_bf16.h>

typedef __attribute__((ext_vector_type(16))) _Float16 v16h;
typedef __attribute__((ext_vector_type(8)))  _Float16 v8h;
typedef __attribute__((ext_vector_type(8)))  float    v8f;

union V16 { v16h v; v8h h[2]; };

#define NROW 32      // N = 2*K rows in similarity matrix
#define NPIX 1024    // H*W
#define NCH  128     // C

// ---------------------------------------------------------------------------
// Kernel 1: softmax over C for each (n,p) row; emit f16 Zh[p][n][c] (pixel-
// major so WMMA fragments are contiguous 16B chunks) + f32 squared norms.
// Lane dim == p  ->  every global load of base[c*1024] is a fully coalesced
// 128B wave transaction.
// ---------------------------------------------------------------------------
__global__ __launch_bounds__(256) void softmax_pack(
    const float* __restrict__ zi1, const float* __restrict__ zi2,
    const float* __restrict__ zj1, const float* __restrict__ zj2,
    _Float16* __restrict__ Zh, float* __restrict__ norm2) {
  int tid = blockIdx.x * 256 + threadIdx.x;      // 0 .. 32767
  int p = tid & (NPIX - 1);
  int n = tid >> 10;                             // uniform per wave (1024 % 32 == 0)
  const float* src = (n < 16) ? ((n < 8) ? zi1 : zi2)
                              : ((n < 24) ? zj1 : zj2);
  const float* base = src + (size_t)(n & 7) * (NCH * NPIX) + p;

  float v[NCH];
  float m = -__builtin_inff();
#pragma unroll
  for (int c = 0; c < NCH; ++c) { v[c] = base[(size_t)c * NPIX]; m = fmaxf(m, v[c]); }
  float s = 0.f;
#pragma unroll
  for (int c = 0; c < NCH; ++c) { v[c] = __expf(v[c] - m); s += v[c]; }
  float inv = 1.f / s;

  float nrm = 0.f;
  _Float16* out = Zh + ((size_t)p * NROW + n) * NCH;
#pragma unroll
  for (int k = 0; k < NCH / 8; ++k) {
    v8h chunk;
#pragma unroll
    for (int e = 0; e < 8; ++e) {
      float q = v[k * 8 + e] * inv;
      nrm += q * q;
      chunk[e] = (_Float16)q;
    }
    *(v8h*)(out + k * 8) = chunk;                // global_store_b128
  }
  norm2[p * NROW + n] = nrm;
}

// ---------------------------------------------------------------------------
// Kernel 2: zero the 32x32 dist accumulator (fresh every launch).
// ---------------------------------------------------------------------------
__global__ void zero_dist(float* __restrict__ dist) {
  dist[threadIdx.x] = 0.f;
}

// ---------------------------------------------------------------------------
// Kernel 3: per-pixel 32x32 Gram via V_WMMA_F32_16X16X32_F16, convert to
// pairwise L2 distance, accumulate sum over pixels with f32 global atomics.
//
// Operand layouts per CDNA5 ISA 7.12.2 (wave32):
//   A 16x32 f16:  lane l<16 holds M=l, K {0..7, 16..23}; lane 16+l holds
//                 M=l, K {8..15, 24..31}  -> two contiguous 16B chunks.
//   B 32x16 f16:  lane l<16 holds N=l, K {0..7, 16..23}; lane 16+l holds N=l,
//                 K {8..15, 24..31}  -> two contiguous 16B chunks.
//   C/D 16x16 f32: lane l, vgpr v -> (M = v + 8*(l>=16), N = l%16).
// Zh is [p][n][c] so every fragment chunk is a direct global_load_b128.
// ---------------------------------------------------------------------------
__global__ __launch_bounds__(128) void gram_dist(
    const _Float16* __restrict__ Zh, const float* __restrict__ norm2,
    float* __restrict__ dist) {
  const int lane = threadIdx.x & 31;
  const int wave = blockIdx.x * 4 + (threadIdx.x >> 5);   // 0 .. 511
  const int half = lane >> 4;                             // 0 | 1
  const int l15  = lane & 15;

  v8f distAcc[2][2];
#pragma unroll
  for (int I = 0; I < 2; ++I)
#pragma unroll
    for (int J = 0; J < 2; ++J) distAcc[I][J] = (v8f)0.f;

#pragma unroll
  for (int pi = 0; pi < 2; ++pi) {
    const int p = wave * 2 + pi;
    const _Float16* zp = Zh + (size_t)p * (NROW * NCH);
    __builtin_prefetch(zp + NROW * NCH, 0, 1);            // global_prefetch_b8 (next pixel)

    v8f g[2][2];
#pragma unroll
    for (int I = 0; I < 2; ++I)
#pragma unroll
      for (int J = 0; J < 2; ++J) g[I][J] = (v8f)0.f;

#pragma unroll
    for (int kk = 0; kk < 4; ++kk) {
      const int k0 = kk * 32;
      V16 a[2], b[2];
#pragma unroll
      for (int I = 0; I < 2; ++I) {
        const _Float16* rowA = zp + (I * 16 + l15) * NCH + k0 + half * 8;
        a[I].h[0] = *(const v8h*)(rowA);                  // K +0..7   (rel)
        a[I].h[1] = *(const v8h*)(rowA + 16);             // K +16..23 (rel)
        const _Float16* rowB = zp + (I * 16 + l15) * NCH + k0 + half * 8;
        b[I].h[0] = *(const v8h*)(rowB);                  // K +0..7   (rel)
        b[I].h[1] = *(const v8h*)(rowB + 16);             // K +16..23 (rel)
      }
      g[0][0] = __builtin_amdgcn_wmma_f32_16x16x32_f16(false, a[0].v, false, b[0].v,
                                                       (short)0, g[0][0], false, false);
      g[0][1] = __builtin_amdgcn_wmma_f32_16x16x32_f16(false, a[0].v, false, b[1].v,
                                                       (short)0, g[0][1], false, false);
      g[1][0] = __builtin_amdgcn_wmma_f32_16x16x32_f16(false, a[1].v, false, b[0].v,
                                                       (short)0, g[1][0], false, false);
      g[1][1] = __builtin_amdgcn_wmma_f32_16x16x32_f16(false, a[1].v, false, b[1].v,
                                                       (short)0, g[1][1], false, false);
    }

    const float* n2 = norm2 + p * NROW;
#pragma unroll
    for (int I = 0; I < 2; ++I) {
      float n2i[8];
#pragma unroll
      for (int e = 0; e < 8; ++e) n2i[e] = n2[I * 16 + half * 8 + e];
#pragma unroll
      for (int J = 0; J < 2; ++J) {
        const float n2j = n2[J * 16 + l15];
#pragma unroll
        for (int e = 0; e < 8; ++e) {
          float d2 = n2i[e] + n2j - 2.0f * g[I][J][e];
          distAcc[I][J][e] += (d2 > 0.f) ? sqrtf(d2) : 0.f;   // matches safe-sqrt
        }
      }
    }
  }

  // Fold this wave's partial sums into the global 32x32 accumulator.
#pragma unroll
  for (int I = 0; I < 2; ++I)
#pragma unroll
    for (int J = 0; J < 2; ++J)
#pragma unroll
      for (int e = 0; e < 8; ++e) {
        int i = I * 16 + half * 8 + e;
        int j = J * 16 + l15;
        __hip_atomic_fetch_add(&dist[i * NROW + j], distAcc[I][J][e],
                               __ATOMIC_RELAXED, __HIP_MEMORY_SCOPE_AGENT);
      }
}

// ---------------------------------------------------------------------------
// Kernel 4: NT-Xent reduction. sim = exp(-dist/P); per row logsumexp over all
// j != i minus the positive logit (equivalent to the masked [pos|neg] layout
// since the positive re-enters the set); sum / N.
// ---------------------------------------------------------------------------
__global__ __launch_bounds__(32) void finalize(const float* __restrict__ dist,
                                               float* __restrict__ out) {
  const int i = threadIdx.x;                    // row 0..31
  const float invP = 1.0f / (float)NPIX;
  float sim[NROW];
#pragma unroll
  for (int j = 0; j < NROW; ++j) sim[j] = __expf(-dist[i * NROW + j] * invP);

  const int pos = (i + 16) & 31;
  float m = -__builtin_inff();
#pragma unroll
  for (int j = 0; j < NROW; ++j) if (j != i) m = fmaxf(m, sim[j]);
  float s = 0.f;
#pragma unroll
  for (int j = 0; j < NROW; ++j) if (j != i) s += __expf(sim[j] - m);
  float contrib = m + __logf(s) - sim[pos];

  for (int off = 16; off > 0; off >>= 1)
    contrib += __shfl_xor(contrib, off, 32);
  if (i == 0) out[0] = contrib * (1.0f / (float)NROW);
}

// ---------------------------------------------------------------------------
// Workspace map:  [0, 8MB)  Zh f16 [1024][32][128]
//                 [8MB, +128KB) norm2 f32 [1024][32]
//                 then dist f32 [32][32]
// ---------------------------------------------------------------------------
extern "C" void kernel_launch(void* const* d_in, const int* in_sizes, int n_in,
                              void* d_out, int out_size, void* d_ws, size_t ws_size,
                              hipStream_t stream) {
  const float* zi1 = (const float*)d_in[0];
  const float* zi2 = (const float*)d_in[1];
  const float* zj1 = (const float*)d_in[2];
  const float* zj2 = (const float*)d_in[3];

  char* ws = (char*)d_ws;
  _Float16* Zh   = (_Float16*)ws;
  float*    nrm2 = (float*)(ws + (size_t)NPIX * NROW * NCH * sizeof(_Float16));
  float*    dist = (float*)((char*)nrm2 + (size_t)NPIX * NROW * sizeof(float));
  float*    out  = (float*)d_out;

  softmax_pack<<<128, 256, 0, stream>>>(zi1, zi2, zj1, zj2, Zh, nrm2);
  zero_dist  <<<1, NROW * NROW, 0, stream>>>(dist);
  gram_dist  <<<128, 128, 0, stream>>>(Zh, nrm2, dist);
  finalize   <<<1, 32, 0, stream>>>(dist, out);
}
